// ImprovedGenerator_46359876993737
// MI455X (gfx1250) — compile-verified, matrix operand order checked by reading, not run on hardware
//
#include <hip/hip_runtime.h>
#include <hip/hip_bf16.h>

// ---------------------------------------------------------------------------
// LSTM stack (dense proj -> 3x LSTM (+residual) -> LayerNorm -> MLP head)
// for MI455X / gfx1250.
//   * GEMMs: v_wmma_f32_16x16x32_f16, 64x128 LDS tiles, double-buffered with
//     GLOBAL_LOAD_ASYNC_TO_LDS_B128 (ASYNCcnt) so copy overlaps compute.
//   * Recurrence: 4 batch-groups x 16 gate-column slices = 64 WGs.
//     512x128 f16 R-slice resident in LDS (128KB of 320KB WGP LDS);
//     h exchange staged via async-to-LDS; per-step sync across the 16 WGs of
//     a batch group via device-scope release/acquire atomic + s_sleep spin.
// ---------------------------------------------------------------------------

typedef __attribute__((ext_vector_type(16))) _Float16 v16h;
typedef __attribute__((ext_vector_type(8)))  float    v8f;
typedef __attribute__((ext_vector_type(4)))  int      v4i;

#define AS1 __attribute__((address_space(1)))
#define AS3 __attribute__((address_space(3)))

#define BB 64
#define TT 256
#define ZZ 128
#define HH 512
#define FF 64
#define BT (BB*TT)      // 16384 rows
#define G4 (4*HH)       // 2048 gate columns

static __device__ __forceinline__ float sigf(float x){ return 1.0f/(1.0f + __expf(-x)); }

// async copy of 16 bytes: global -> LDS via the async DMA path (ASYNCcnt)
static __device__ __forceinline__ void async_cp16(const _Float16* gsrc, _Float16* lds_dst){
  __builtin_amdgcn_global_load_async_to_lds_b128(
      (AS1 v4i*)gsrc, (AS3 v4i*)lds_dst, 0, 0);
}
static __device__ __forceinline__ void wait_async0(){
  asm volatile("s_wait_asynccnt 0" ::: "memory");
}

// ------------------------------ elementwise --------------------------------
__global__ void k_cvt(const float* __restrict__ x, _Float16* __restrict__ y, int n){
  int i = blockIdx.x*256 + threadIdx.x;
  if (i < n) y[i] = (_Float16)x[i];
}

// W[K][N] fp32 -> Wt[N][K] f16 (so B-fragments are contiguous along K)
__global__ void k_transpose(const float* __restrict__ W, _Float16* __restrict__ Wt, int K, int N){
  int i = blockIdx.x*256 + threadIdx.x;
  if (i < K*N){ int k = i / N, n = i - k*N; Wt[(size_t)n*K + k] = (_Float16)W[i]; }
}

__global__ void k_add(const _Float16* __restrict__ a, const _Float16* __restrict__ b,
                      _Float16* __restrict__ c, int n){
  int i = blockIdx.x*256 + threadIdx.x;
  if (i < n) c[i] = (_Float16)((float)a[i] + (float)b[i]);
}

__global__ void k_zero_u32(unsigned* p, int n){
  int i = blockIdx.x*256 + threadIdx.x;
  if (i < n) p[i] = 0u;
}

// ------------------------------ WMMA GEMM ----------------------------------
// C[M,N] = act(A[M,K] @ Bt[N,K]^T + bias).  M%64==0, N%BN==0, K%32==0.
// 256 threads = 8 waves; wave = (mi 0..3) x (ni 0..1); wave tile 16 x (BN/2).
// K staged in 32-wide chunks, double-buffered via async global->LDS copies.
template<int BN>
__device__ __forceinline__ void stage_async(const _Float16* A, const _Float16* Bt,
                                            _Float16* As, _Float16* Bs,
                                            int rowBase, int colBase, int K, int k0, int tid)
{
  const int ar = tid >> 2, aseg = (tid & 3) * 8;
  async_cp16(A + (size_t)(rowBase + ar)*K + k0 + aseg, As + ar*32 + aseg);
#pragma unroll
  for (int i = 0; i < BN/64; ++i){
    int idx = tid + i*256;
    int c = idx >> 2, seg = (idx & 3) * 8;
    async_cp16(Bt + (size_t)(colBase + c)*K + k0 + seg, Bs + c*32 + seg);
  }
}

template<int BN>
__global__ __launch_bounds__(256)
void k_gemm(const _Float16* __restrict__ A, const _Float16* __restrict__ Bt,
            const float* __restrict__ bias, _Float16* __restrict__ outH,
            float* __restrict__ outF, int M, int N, int K, int act)
{
  constexpr int NSUB = BN / 32;             // 16-wide wmma tiles per wave
  __shared__ _Float16 As[2][64*32];
  __shared__ _Float16 Bs[2][BN*32];
  const int tid  = threadIdx.x;
  const int wave = tid >> 5, lane = tid & 31;
  const int mi = wave & 3, ni = wave >> 2;
  const int rowBase = blockIdx.x*64, colBase = blockIdx.y*BN;
  const int hi8 = (lane >= 16) ? 8 : 0;     // K-offset per 16-bit A/B layout
  const int lm  = lane & 15;

  v8f zero = {0,0,0,0,0,0,0,0};
  v8f acc[NSUB];
#pragma unroll
  for (int s = 0; s < NSUB; ++s) acc[s] = zero;

  stage_async<BN>(A, Bt, As[0], Bs[0], rowBase, colBase, K, 0, tid);
  wait_async0();
  __syncthreads();

  const int nk = K >> 5;
  for (int kc = 0; kc < nk; ++kc){
    const int cur = kc & 1;
    if (kc + 1 < nk)
      stage_async<BN>(A, Bt, As[cur^1], Bs[cur^1], rowBase, colBase, K, (kc+1)*32, tid);

    union { v16h v; uint4 q[2]; } ua;
    const _Float16* ap = &As[cur][(mi*16 + lm)*32 + hi8];
    ua.q[0] = *(const uint4*)ap; ua.q[1] = *(const uint4*)(ap + 16);
#pragma unroll
    for (int s = 0; s < NSUB; ++s){
      union { v16h v; uint4 q[2]; } ub;
      const _Float16* bp = &Bs[cur][(ni*(BN/2) + s*16 + lm)*32 + hi8];
      ub.q[0] = *(const uint4*)bp; ub.q[1] = *(const uint4*)(bp + 16);
      acc[s] = __builtin_amdgcn_wmma_f32_16x16x32_f16(false, ua.v, false, ub.v,
                                                      (short)0, acc[s], false, false);
    }
    wait_async0();          // next buffer fully written (per-wave)
    __syncthreads();        // all waves done reading cur + next buffer ready
  }

#pragma unroll
  for (int s = 0; s < NSUB; ++s){
    int col = colBase + ni*(BN/2) + s*16 + lm;
    float bv = bias ? bias[col] : 0.0f;
#pragma unroll
    for (int rr = 0; rr < 8; ++rr){
      int row = rowBase + mi*16 + rr + hi8;
      float v = acc[s][rr] + bv;
      if (act == 1)      v = fmaxf(v, 0.0f);
      else if (act == 2) v = tanhf(v);
      if (outF) outF[(size_t)row*N + col] = v;
      else      outH[(size_t)row*N + col] = (_Float16)v;
    }
  }
}

// ------------------------------ LSTM scan ----------------------------------
// grid = (4 batch groups, 16 column slices), block = 64 threads (2 waves).
// Rt: [2048 gate cols][512] f16 (transposed recurrent kernel)
// xK: [B][T][2048] f16 precomputed x@K + b
// hbuf: [2][4][16*512] f16 double-buffered h exchange, cnt: [4] barrier ctr
__global__ __launch_bounds__(64)
void k_lstm(const _Float16* __restrict__ Rt, const _Float16* __restrict__ xK,
            _Float16* __restrict__ hbuf, unsigned* __restrict__ cnt,
            _Float16* __restrict__ Y)
{
  extern __shared__ __align__(16) char smem[];
  _Float16* Rs = (_Float16*)smem;                                     // [128][512]
  _Float16* Hs = (_Float16*)(smem + (size_t)128*HH*sizeof(_Float16)); // [16][512]
  const int bg = blockIdx.x, ns = blockIdx.y;
  const int tid  = threadIdx.x;
  const int wave = tid >> 5, lane = tid & 31;
  const int j0 = ns * 32;            // this WG's 32 h-columns
  const int jw = j0 + wave*16;       // this wave's 16 h-columns
  const int lm  = lane & 15;
  const int hi8 = (lane >= 16) ? 8 : 0;

  // Cache R slice (i,f,g,o blocks of 32 cols each) into LDS via async DMA.
  for (int idx = tid; idx < 128*64; idx += 64){
    int c = idx >> 6, q = idx & 63;
    int gcol = (c >> 5)*HH + j0 + (c & 31);
    async_cp16(Rt + (size_t)gcol*HH + q*8, Rs + (size_t)c*HH + q*8);
  }
  wait_async0();
  __syncthreads();

  v8f cst = {0,0,0,0,0,0,0,0};   // cell state, WMMA C/D layout

  for (int t = 0; t < TT; ++t){
    if (t > 0){  // stage h_{t-1} (16x512) written by all 16 WGs of this group
      const _Float16* hp = hbuf + (size_t)(((t & 1) ^ 1)*4 + bg)*16*HH;
      for (int i = tid; i < 16*HH/8; i += 64)
        async_cp16(hp + (size_t)i*8, Hs + (size_t)i*8);
      wait_async0();
    }
    __syncthreads();

    // init accumulators from precomputed xK (+bias folded in)
    v8f acc[4];
    const _Float16* xbase = xK + ((size_t)(bg*16)*TT + t)*G4;
#pragma unroll
    for (int gi = 0; gi < 4; ++gi){
      const _Float16* xp = xbase + gi*HH + jw + lm;
      v8f a;
#pragma unroll
      for (int rr = 0; rr < 8; ++rr)
        a[rr] = (float)xp[(size_t)(rr + hi8)*TT*G4];
      acc[gi] = a;
    }
    if (t + 1 < TT) __builtin_prefetch(xbase + G4 + jw, 0, 1);

    if (t > 0){  // h_{t-1} @ R-slice : 16 K-chunks x 4 gates of WMMA
      for (int kc = 0; kc < 16; ++kc){
        union { v16h v; uint4 q[2]; } ua;
        const _Float16* ap = Hs + lm*HH + kc*32 + hi8;
        ua.q[0] = *(const uint4*)ap; ua.q[1] = *(const uint4*)(ap + 16);
#pragma unroll
        for (int gi = 0; gi < 4; ++gi){
          union { v16h v; uint4 q[2]; } ub;
          const _Float16* bp = Rs + (size_t)(gi*32 + wave*16 + lm)*HH + kc*32 + hi8;
          ub.q[0] = *(const uint4*)bp; ub.q[1] = *(const uint4*)(bp + 16);
          acc[gi] = __builtin_amdgcn_wmma_f32_16x16x32_f16(false, ua.v, false, ub.v,
                                                           (short)0, acc[gi], false, false);
        }
      }
    }

    // gates (keras order i,f,g,o), cell/h update, write h slice
    _Float16* hw = hbuf + (size_t)((t & 1)*4 + bg)*16*HH;
#pragma unroll
    for (int rr = 0; rr < 8; ++rr){
      float iv = sigf(acc[0][rr]);
      float fv = sigf(acc[1][rr]);
      float gv = tanhf(acc[2][rr]);
      float ov = sigf(acc[3][rr]);
      float c  = fv*cst[rr] + iv*gv;
      cst[rr]  = c;
      _Float16 hh = (_Float16)(ov*tanhf(c));
      int m = rr + hi8;
      hw[m*HH + jw + lm] = hh;
      Y[((size_t)(bg*16 + m)*TT + t)*HH + jw + lm] = hh;
    }

    // device-scope arrive/spin barrier across the 16 WGs of this batch group
    __threadfence();
    __syncthreads();
    if (tid == 0){
      __hip_atomic_fetch_add(cnt + bg, 1u, __ATOMIC_RELEASE, __HIP_MEMORY_SCOPE_AGENT);
      const unsigned target = 16u*(unsigned)(t + 1);
      while (__hip_atomic_load(cnt + bg, __ATOMIC_ACQUIRE, __HIP_MEMORY_SCOPE_AGENT) < target)
        __builtin_amdgcn_s_sleep(2);
    }
    __syncthreads();
  }
}

// ------------------------------ LayerNorm ----------------------------------
__global__ __launch_bounds__(256)
void k_layernorm(const _Float16* __restrict__ X, const float* __restrict__ gamma,
                 const float* __restrict__ beta, _Float16* __restrict__ Yo)
{
  const int wave = threadIdx.x >> 5, lane = threadIdx.x & 31;
  const int row = blockIdx.x*8 + wave;
  const _Float16* xp = X + (size_t)row*HH;
  float vals[16];
  float s = 0.f, sq = 0.f;
#pragma unroll
  for (int i = 0; i < 16; ++i){
    float v = (float)xp[lane + i*32];
    vals[i] = v; s += v; sq += v*v;
  }
#pragma unroll
  for (int m = 16; m >= 1; m >>= 1){
    s  += __shfl_xor(s,  m, 32);
    sq += __shfl_xor(sq, m, 32);
  }
  float mu   = s * (1.0f/HH);
  float var  = sq * (1.0f/HH) - mu*mu;
  float rstd = rsqrtf(var + 1e-3f);
  _Float16* yp = Yo + (size_t)row*HH;
#pragma unroll
  for (int i = 0; i < 16; ++i){
    int j = lane + i*32;
    yp[j] = (_Float16)((vals[i] - mu)*rstd*gamma[j] + beta[j]);
  }
}

// ------------------------------ driver -------------------------------------
extern "C" void kernel_launch(void* const* d_in, const int* in_sizes, int n_in,
                              void* d_out, int out_size, void* d_ws, size_t ws_size,
                              hipStream_t stream)
{
  (void)in_sizes; (void)n_in; (void)out_size; (void)ws_size;
  const float* z   = (const float*)d_in[0];
  const float* Wp  = (const float*)d_in[1];
  const float* bp  = (const float*)d_in[2];
  const float* Ws  = (const float*)d_in[3];
  const float* bs  = (const float*)d_in[4];
  const float* K0  = (const float*)d_in[5];
  const float* R0  = (const float*)d_in[6];
  const float* b0  = (const float*)d_in[7];
  const float* K1  = (const float*)d_in[8];
  const float* R1  = (const float*)d_in[9];
  const float* b1  = (const float*)d_in[10];
  const float* K2  = (const float*)d_in[11];
  const float* R2  = (const float*)d_in[12];
  const float* b2  = (const float*)d_in[13];
  const float* gamma = (const float*)d_in[14];
  const float* beta  = (const float*)d_in[15];
  const float* W1  = (const float*)d_in[16];
  const float* b1d = (const float*)d_in[17];
  const float* W2  = (const float*)d_in[18];
  const float* b2d = (const float*)d_in[19];

  char* ws = (char*)d_ws;
  size_t off = 0;
  auto take = [&](size_t bytes)->char* {
    char* p = ws + off; off = (off + bytes + 255) & ~(size_t)255; return p;
  };
  _Float16* zh   = (_Float16*)take((size_t)BT*ZZ*2);
  _Float16* Wpt  = (_Float16*)take((size_t)(2*HH)*ZZ*2);
  _Float16* Wst  = (_Float16*)take((size_t)HH*ZZ*2);
  _Float16* K0t  = (_Float16*)take((size_t)G4*(2*HH)*2);
  _Float16* K1t  = (_Float16*)take((size_t)G4*HH*2);
  _Float16* K2t  = (_Float16*)take((size_t)G4*HH*2);
  _Float16* R0t  = (_Float16*)take((size_t)G4*HH*2);
  _Float16* R1t  = (_Float16*)take((size_t)G4*HH*2);
  _Float16* R2t  = (_Float16*)take((size_t)G4*HH*2);
  _Float16* W1t  = (_Float16*)take((size_t)HH*HH*2);
  _Float16* W2t  = (_Float16*)take((size_t)FF*HH*2);
  _Float16* xh   = (_Float16*)take((size_t)BT*(2*HH)*2);   // dense proj out
  _Float16* res  = (_Float16*)take((size_t)BT*HH*2);       // skip proj out
  _Float16* xKb  = (_Float16*)take((size_t)BT*G4*2);       // per-layer x@K+b
  _Float16* bufA = (_Float16*)take((size_t)BT*HH*2);       // activation ring
  _Float16* bufB = (_Float16*)take((size_t)BT*HH*2);
  _Float16* hbuf = (_Float16*)take((size_t)2*4*16*HH*2);   // h exchange
  unsigned* cnt  = (unsigned*)take(3*4*sizeof(unsigned));  // barrier counters

  auto cvt = [&](const float* s, _Float16* d, int n){
    k_cvt<<<(n+255)/256, 256, 0, stream>>>(s, d, n);
  };
  auto tr = [&](const float* s, _Float16* d, int K, int N){
    k_transpose<<<((K*N)+255)/256, 256, 0, stream>>>(s, d, K, N);
  };
  auto gemm128 = [&](const _Float16* A, const _Float16* Bt, const float* bias,
                     _Float16* oH, float* oF, int M, int N, int K, int act){
    dim3 g(M/64, N/128);
    k_gemm<128><<<g, 256, 0, stream>>>(A, Bt, bias, oH, oF, M, N, K, act);
  };
  auto gemm64 = [&](const _Float16* A, const _Float16* Bt, const float* bias,
                    _Float16* oH, float* oF, int M, int N, int K, int act){
    dim3 g(M/64, N/64);
    k_gemm<64><<<g, 256, 0, stream>>>(A, Bt, bias, oH, oF, M, N, K, act);
  };

  k_zero_u32<<<1, 32, 0, stream>>>(cnt, 12);
  cvt(z, zh, BT*ZZ);
  tr(Wp, Wpt, ZZ, 2*HH);
  tr(Ws, Wst, ZZ, HH);
  tr(K0, K0t, 2*HH, G4);
  tr(K1, K1t, HH, G4);
  tr(K2, K2t, HH, G4);
  tr(R0, R0t, HH, G4);
  tr(R1, R1t, HH, G4);
  tr(R2, R2t, HH, G4);
  tr(W1, W1t, HH, HH);
  tr(W2, W2t, HH, FF);

  // dense / skip projections
  gemm128(zh, Wpt, bp, xh,  nullptr, BT, 2*HH, ZZ, 0);
  gemm128(zh, Wst, bs, res, nullptr, BT, HH,   ZZ, 0);

  const dim3 sg(4, 16);
  const size_t lds = ((size_t)128*HH + (size_t)16*HH) * sizeof(_Float16); // 144 KiB

  // layer 0
  gemm128(xh, K0t, b0, xKb, nullptr, BT, G4, 2*HH, 0);
  k_lstm<<<sg, 64, lds, stream>>>(R0t, xKb, hbuf, cnt + 0, bufA);
  // layer 1
  gemm128(bufA, K1t, b1, xKb, nullptr, BT, G4, HH, 0);
  k_lstm<<<sg, 64, lds, stream>>>(R1t, xKb, hbuf, cnt + 4, bufB);
  // residual add (after layer index 1)
  k_add<<<(BT*HH+255)/256, 256, 0, stream>>>(bufB, res, bufA, BT*HH);
  // layer 2
  gemm128(bufA, K2t, b2, xKb, nullptr, BT, G4, HH, 0);
  k_lstm<<<sg, 64, lds, stream>>>(R2t, xKb, hbuf, cnt + 8, bufB);

  // LayerNorm + MLP head
  k_layernorm<<<BT/8, 256, 0, stream>>>(bufB, gamma, beta, bufA);
  gemm128(bufA, W1t, b1d, bufB, nullptr, BT, HH, HH, 1);           // ReLU
  gemm64(bufB, W2t, b2d, nullptr, (float*)d_out, BT, FF, HH, 2);   // tanh -> out
}
